// EncoderDecoderRetrievalModel_85409719648608
// MI455X (gfx1250) — matrix operand order repaired and checked by reading, not compiled, not run on hardware
//
#include <hip/hip_runtime.h>
#include <hip/hip_bf16.h>

// ---------------------------------------------------------------------------
// EncoderDecoderRetrievalModel forward for MI455X (gfx1250, wave32, WMMA).
//
// Heavy path (fused): full-table _proj(text)/_proj(image) + 3-way scoring,
// one pass over the 205MB tables, f16 WMMA with f32 accumulation.
// Weights pre-packed into WMMA B-fragment order -> all B loads are b128.
// Fast transcendentals (v_exp_f32 / v_rcp_f32) for gelu/softmax.
// ---------------------------------------------------------------------------

typedef _Float16 f16;
typedef __attribute__((ext_vector_type(16))) _Float16 v16h;
typedef __attribute__((ext_vector_type(4)))  _Float16 v4h;
typedef __attribute__((ext_vector_type(8)))  float    v8f;

#define BB 64
#define LL 200
#define DD 64
#define NITEMS 50001          // N+1 rows (row 50000 = padding)
#define PADID 50000
#define BL (BB*LL)            // 12800 rows (multiple of 16)
#define TEMP_INV (1.0f/0.07f)

// ------------------------------ WMMA helpers -------------------------------

__device__ __forceinline__ v8f wmma16(v16h a, v16h b, v8f c) {
  return __builtin_amdgcn_wmma_f32_16x16x32_f16(false, a, false, b, (short)0, c,
                                                false, false);
}

// A fragment (16x32, MxK) from row-major f16 (LDS). lane: m=lane&15, half=lane>>4.
// vgpr v: k = k0 + half*16 + v*2 (+slot) -> one 32B run.
__device__ __forceinline__ v16h load_a_lds(const f16* base, int stride, int k0, int lane) {
  v16h a;
  int m = lane & 15, half = lane >> 4;
#pragma unroll
  for (int v = 0; v < 8; ++v) {
    int kb = k0 + half * 16 + v * 2;
    a[2 * v]     = base[m * stride + kb];
    a[2 * v + 1] = base[m * stride + kb + 1];
  }
  return a;
}

// Packed-fragment loads: one lane's 16 f16 stored contiguously (32B).
__device__ __forceinline__ v16h load_frag(const f16* packed, int frag, int lane) {
  return *(const v16h*)(packed + (((size_t)frag * 32 + lane) << 4));
}

// fast gelu (tanh approx): tanh(u) = 1 - 2/(exp(2u)+1) via v_exp/v_rcp
__device__ __forceinline__ float gelu_f(float x) {
  const float c = 0.7978845608028654f;
  float u = c * (x + 0.044715f * x * x * x);
  float t = __expf(2.0f * u);
  float th = 1.0f - 2.0f * __builtin_amdgcn_rcpf(t + 1.0f);
  return 0.5f * x * (1.0f + th);
}

// ------------------------ weight pre-pack (B fragments) --------------------
// W[K x N] f32 -> packed f16: fragment (kt,nt), lane, 16 contiguous values.
// j -> k = kt*32 + half*16 + j, n = nt*16 + (lane&15).
__global__ void pack_b_kernel(const float* __restrict__ W, f16* __restrict__ Wp,
                              int K, int N) {
  int t = blockIdx.x * 256 + threadIdx.x;
  int total = (K * N) >> 4;
  if (t >= total) return;
  int frag = t >> 5, lane = t & 31;
  int ntiles = N >> 4;
  int kt = frag / ntiles, nt = frag - kt * ntiles;
  int n = nt * 16 + (lane & 15);
  int half = lane >> 4;
  v16h pk;
#pragma unroll
  for (int j = 0; j < 16; ++j) {
    int k = kt * 32 + half * 16 + j;
    pk[j] = (f16)W[(size_t)k * N + n];
  }
  *(v16h*)(Wp + ((size_t)t << 4)) = pk;
}

// --------------------------- generic row GEMM ------------------------------
// C[M x N] = epi(A[M x K] @ W + bias), W pre-packed. M multiple of 16.
// Optional rowidx (gather). EPI: 0=none, 1=gelu.
template <int K, int N, int EPI>
__global__ __launch_bounds__(128) void gemm_kernel(
    const float* __restrict__ A, const int* __restrict__ rowidx,
    const f16* __restrict__ Wp, const float* __restrict__ bias,
    float* __restrict__ C) {
  __shared__ f16 As[16 * K];
  int tid = threadIdx.x;
  int blk = blockIdx.x;
  constexpr int NT = N / 16;
  // stage 16 rows of A -> f16 LDS (float4 in, v4h out)
  for (int e = tid; e < 4 * K; e += 128) {
    int r = e / (K / 4), c4 = e - r * (K / 4);
    int gr = blk * 16 + r;
    int ar = rowidx ? rowidx[gr] : gr;
    float4 f4 = ((const float4*)(A + (size_t)ar * K))[c4];
    v4h h; h[0] = (f16)f4.x; h[1] = (f16)f4.y; h[2] = (f16)f4.z; h[3] = (f16)f4.w;
    *(v4h*)(As + r * K + c4 * 4) = h;
  }
  __syncthreads();
  int w = tid >> 5, lane = tid & 31, half = lane >> 4, nl = lane & 15;
  for (int nt = w; nt < NT; nt += 4) {
    v8f acc;
#pragma unroll
    for (int i = 0; i < 8; ++i) acc[i] = 0.0f;
    for (int kt = 0; kt < K / 32; ++kt) {
      v16h a = load_a_lds(As, K, kt * 32, lane);
      v16h b = load_frag(Wp, kt * NT + nt, lane);
      acc = wmma16(a, b, acc);
    }
#pragma unroll
    for (int v = 0; v < 8; ++v) {
      int m = v + 8 * half;
      int n = nt * 16 + nl;
      float val = acc[v] + bias[n];
      if (EPI == 1) val = gelu_f(val);
      C[(size_t)(blk * 16 + m) * N + n] = val;
    }
  }
}

// --------------------------- small utility kernels -------------------------

__global__ void lens_kernel(const int* __restrict__ hist, int* __restrict__ lens) {
  int b = threadIdx.x;
  if (b < BB) {
    int c = 0;
    for (int l = 0; l < LL; ++l) c += (hist[b * LL + l] != PADID) ? 1 : 0;
    lens[b] = c;
  }
}

__global__ void gather_item_kernel(const int* __restrict__ hist,
                                   const float* __restrict__ emb,
                                   float* __restrict__ out) {
  int t = blockIdx.x * 256 + threadIdx.x;
  if (t < BL * DD) {
    int r = t >> 6, d = t & 63;
    out[t] = emb[(size_t)hist[r] * DD + d];
  }
}

// out = LN( x + s*y + pos ),  s = (*scale) * gate[row,gcomp]
__global__ __launch_bounds__(64) void combine_ln_kernel(
    float* __restrict__ out, const float* __restrict__ x,
    const float* __restrict__ y, const float* __restrict__ scale_p,
    const float* __restrict__ gate, int gcomp,
    const float* __restrict__ pos, int lmod,
    const float* __restrict__ g, const float* __restrict__ bb, float eps) {
  int row = blockIdx.x, d = threadIdx.x;
  float s = 1.0f;
  if (scale_p) s *= *scale_p;
  if (gate) s *= gate[row * 2 + gcomp];
  float v = x[(size_t)row * DD + d];
  if (y) v += s * y[(size_t)row * DD + d];
  if (pos) v += pos[(size_t)(row % lmod) * DD + d];
  __shared__ float red[64];
  red[d] = v; __syncthreads();
  for (int st = 32; st > 0; st >>= 1) { if (d < st) red[d] += red[d + st]; __syncthreads(); }
  float mean = red[0] * (1.0f / 64.0f); __syncthreads();
  float df = v - mean;
  red[d] = df * df; __syncthreads();
  for (int st = 32; st > 0; st >>= 1) { if (d < st) red[d] += red[d + st]; __syncthreads(); }
  float var = red[0] * (1.0f / 64.0f); __syncthreads();
  out[(size_t)row * DD + d] = df * rsqrtf(var + eps) * g[d] + bb[d];
}

// gate = softmax over 2 of (gh @ w + b)   (gh: BL x 32)
__global__ void gate_kernel(const float* __restrict__ gh, const float* __restrict__ w,
                            const float* __restrict__ b, float* __restrict__ gate) {
  int r = blockIdx.x * 256 + threadIdx.x;
  if (r < BL) {
    float l0 = b[0], l1 = b[1];
    for (int k = 0; k < 32; ++k) {
      float h = gh[r * 32 + k];
      l0 += h * w[k * 2]; l1 += h * w[k * 2 + 1];
    }
    float mx = fmaxf(l0, l1);
    float e0 = __expf(l0 - mx), e1 = __expf(l1 - mx);
    float s = __builtin_amdgcn_rcpf(e0 + e1);
    gate[r * 2] = e0 * s; gate[r * 2 + 1] = e1 * s;
  }
}

// MHA core (online softmax). Q/K/V: (B*L, 64) with head h at feature h*DH.
template <int DH, bool CAUSAL>
__global__ __launch_bounds__(256) void attn_kernel(
    const float* __restrict__ Q, const float* __restrict__ K,
    const float* __restrict__ V, const int* __restrict__ lens,
    float* __restrict__ O, int H) {
  int b = blockIdx.x / H, h = blockIdx.x % H;
  int q = threadIdx.x;
  if (q >= LL) return;
  int len = lens[b];
  const float scale = rsqrtf((float)DH);
  float qv[DH], o[DH];
  int base = (b * LL + q) * DD + h * DH;
#pragma unroll
  for (int d = 0; d < DH; ++d) { qv[d] = Q[base + d]; o[d] = 0.0f; }
  int kl = CAUSAL ? min(q + 1, len) : len;
  float m = -1e30f, s = 0.0f;
  for (int k = 0; k < kl; ++k) {
    const float* kp = K + (b * LL + k) * DD + h * DH;
    float dot = 0.0f;
#pragma unroll
    for (int d = 0; d < DH; ++d) dot += qv[d] * kp[d];
    dot *= scale;
    float nm = fmaxf(m, dot);
    float corr = __expf(m - nm);
    float wk = __expf(dot - nm);
    s = s * corr + wk;
    const float* vp = V + (b * LL + k) * DD + h * DH;
#pragma unroll
    for (int d = 0; d < DH; ++d) o[d] = o[d] * corr + wk * vp[d];
    m = nm;
  }
  float inv = 1.0f / s;
#pragma unroll
  for (int d = 0; d < DH; ++d) O[base + d] = o[d] * inv;
}

// _self_attend core: softmax(Q K^T / 8) V, masked-mean over valid queries.
__global__ __launch_bounds__(256) void selfattend_kernel(
    const float* __restrict__ Q, const float* __restrict__ K,
    const float* __restrict__ V, const int* __restrict__ lens,
    float* __restrict__ pref) {
  int b = blockIdx.x, q = threadIdx.x;
  int len = lens[b];
  __shared__ float acc[64];
  if (threadIdx.x < 64) acc[threadIdx.x] = 0.0f;
  __syncthreads();
  if (q < len) {
    float qv[64], o[64];
    const float* qp = Q + (b * LL + q) * DD;
#pragma unroll
    for (int d = 0; d < 64; ++d) { qv[d] = qp[d]; o[d] = 0.0f; }
    float m = -1e30f, s = 0.0f;
    for (int k = 0; k < len; ++k) {
      const float* kp = K + (b * LL + k) * DD;
      float dot = 0.0f;
#pragma unroll
      for (int d = 0; d < 64; ++d) dot += qv[d] * kp[d];
      dot *= 0.125f;
      float nm = fmaxf(m, dot);
      float corr = __expf(m - nm), wk = __expf(dot - nm);
      s = s * corr + wk;
      const float* vp = V + (b * LL + k) * DD;
#pragma unroll
      for (int d = 0; d < 64; ++d) o[d] = o[d] * corr + wk * vp[d];
      m = nm;
    }
    float inv = 1.0f / s;
#pragma unroll
    for (int d = 0; d < 64; ++d) atomicAdd(&acc[d], o[d] * inv);
  }
  __syncthreads();
  if (threadIdx.x < 64)
    pref[b * 64 + threadIdx.x] = acc[threadIdx.x] / (float)len;
}

// router: h = l1(concat) -> L2-normalize -> relu -> l2 -> softmax(/TEMP)
__global__ void router_kernel(const float* __restrict__ pi, const float* __restrict__ pt,
                              const float* __restrict__ pv,
                              const float* __restrict__ w1, const float* __restrict__ b1,
                              const float* __restrict__ w2, const float* __restrict__ b2,
                              float* __restrict__ mw) {
  int b = threadIdx.x;
  if (b >= BB) return;
  float h[24];
  for (int j = 0; j < 24; ++j) h[j] = b1[j];
  for (int f = 0; f < 64; ++f) {
    float x = pi[b * 64 + f];
    for (int j = 0; j < 24; ++j) h[j] += x * w1[f * 24 + j];
  }
  for (int f = 0; f < 64; ++f) {
    float x = pt[b * 64 + f];
    for (int j = 0; j < 24; ++j) h[j] += x * w1[(64 + f) * 24 + j];
  }
  for (int f = 0; f < 64; ++f) {
    float x = pv[b * 64 + f];
    for (int j = 0; j < 24; ++j) h[j] += x * w1[(128 + f) * 24 + j];
  }
  float nrm = 0.0f;
  for (int j = 0; j < 24; ++j) nrm += h[j] * h[j];
  float inv = 1.0f / fmaxf(sqrtf(nrm), 1e-12f);
  float l[3] = {b2[0], b2[1], b2[2]};
  for (int j = 0; j < 24; ++j) {
    float hv = fmaxf(h[j] * inv, 0.0f);
    for (int c = 0; c < 3; ++c) l[c] += hv * w2[j * 3 + c];
  }
  float mx = fmaxf(l[0], fmaxf(l[1], l[2]));
  float e[3], s = 0.0f;
  for (int c = 0; c < 3; ++c) { e[c] = __expf((l[c] - mx) * TEMP_INV); s += e[c]; }
  for (int c = 0; c < 3; ++c) mw[b * 3 + c] = e[c] / s;
}

// queries, pre-scaled by mw/TEMP, emitted directly in packed A-fragment order.
// frag = md*8 + mt*2 + kt; lane: m = mt*16+(lane&15) = batch; 16 f16/lane.
__global__ void build_queries_kernel(const float* __restrict__ mw, const int* __restrict__ lens,
                                     const float* __restrict__ item_en,
                                     const float* __restrict__ tde2,
                                     const float* __restrict__ vde2,
                                     f16* __restrict__ querP) {
  int t = blockIdx.x * 256 + threadIdx.x;
  if (t >= 3 * 8 * 32) return;
  int md = t >> 8;
  int rem = t & 255;
  int frag = rem >> 5, lane = rem & 31;
  int mt = frag >> 1, kt = frag & 1;
  int b = mt * 16 + (lane & 15);
  int half = lane >> 4;
  const float* src = (md == 0) ? item_en : (md == 1) ? tde2 : vde2;
  int row = b * LL + lens[b] - 1;
  float sc = mw[b * 3 + md] * TEMP_INV;
  v16h pk;
#pragma unroll
  for (int j = 0; j < 16; ++j) {
    int v = j >> 1, slot = j & 1;
    int k = kt * 32 + half * 16 + v * 2 + slot;
    pk[j] = (f16)(sc * src[(size_t)row * 64 + k]);
  }
  *(v16h*)(querP + ((size_t)t << 4)) = pk;
}

// -------------------- fused full-table proj + scoring ----------------------

struct ProjW {
  const f16 *w1p, *w2p, *resp;                 // packed B fragments
  const float *b1, *b2, *rb, *alpha, *g, *b;   // f32 biases / LN params
};

// proj one 64-item tile into Fout (f16, B-fragment packed, post-LayerNorm)
__device__ void proj_tile(const float* __restrict__ table, ProjW p, int i0,
                          f16* Xs, f16* Hs, float* Ps, f16* Fout) {
  int tid = threadIdx.x;
  int w = tid >> 5, lane = tid & 31, half = lane >> 4, nl = lane & 15;
  // stage 64x512 f32 -> f16 LDS
  for (int e = tid; e < 64 * 128; e += 256) {
    int row = e >> 7, c4 = e & 127;
    int item = min(i0 + row, NITEMS - 1);
    float4 f4 = ((const float4*)(table + (size_t)item * 512))[c4];
    v4h hh; hh[0] = (f16)f4.x; hh[1] = (f16)f4.y; hh[2] = (f16)f4.z; hh[3] = (f16)f4.w;
    *(v4h*)(Xs + row * 512 + c4 * 4) = hh;
  }
  __syncthreads();
  // GEMM1: H = gelu(X @ w1 + b1)   (64x512)@(512x128)
  for (int t = w; t < 32; t += 8) {
    int mt = t >> 3, nt = t & 7;
    v8f acc;
#pragma unroll
    for (int i = 0; i < 8; ++i) acc[i] = 0.0f;
    for (int kt = 0; kt < 16; ++kt) {
      v16h a = load_a_lds(Xs + mt * 16 * 512, 512, kt * 32, lane);
      v16h b = load_frag(p.w1p, kt * 8 + nt, lane);
      acc = wmma16(a, b, acc);
    }
#pragma unroll
    for (int v = 0; v < 8; ++v) {
      int m = mt * 16 + v + 8 * half, n = nt * 16 + nl;
      Hs[m * 128 + n] = (f16)gelu_f(acc[v] + p.b1[n]);
    }
  }
  __syncthreads();
  // GEMM2: P = alpha*(H @ w2 + b2) + (X @ res + rb)
  float alpha = *p.alpha;
  for (int t = w; t < 16; t += 8) {
    int mt = t >> 2, nt = t & 3;
    v8f a1, a2;
#pragma unroll
    for (int i = 0; i < 8; ++i) { a1[i] = 0.0f; a2[i] = 0.0f; }
    for (int kt = 0; kt < 4; ++kt) {
      v16h a = load_a_lds(Hs + mt * 16 * 128, 128, kt * 32, lane);
      v16h b = load_frag(p.w2p, kt * 4 + nt, lane);
      a1 = wmma16(a, b, a1);
    }
    for (int kt = 0; kt < 16; ++kt) {
      v16h a = load_a_lds(Xs + mt * 16 * 512, 512, kt * 32, lane);
      v16h b = load_frag(p.resp, kt * 4 + nt, lane);
      a2 = wmma16(a, b, a2);
    }
#pragma unroll
    for (int v = 0; v < 8; ++v) {
      int m = mt * 16 + v + 8 * half, n = nt * 16 + nl;
      Ps[m * 64 + n] = alpha * (a1[v] + p.b2[n]) + a2[v] + p.rb[n];
    }
  }
  __syncthreads();
  // row LayerNorm (eps 1e-5) -> B-fragment-packed f16
  if (tid < 64) {
    int r = tid;
    float mean = 0.0f;
    for (int d = 0; d < 64; ++d) mean += Ps[r * 64 + d];
    mean *= (1.0f / 64.0f);
    float var = 0.0f;
    for (int d = 0; d < 64; ++d) { float df = Ps[r * 64 + d] - mean; var += df * df; }
    var *= (1.0f / 64.0f);
    float inv = rsqrtf(var + 1e-5f);
    int nt = r >> 4, lr = r & 15;
#pragma unroll
    for (int kt = 0; kt < 2; ++kt)
#pragma unroll
      for (int hf = 0; hf < 2; ++hf) {
        v16h pk;
#pragma unroll
        for (int j = 0; j < 16; ++j) {
          int k = kt * 32 + hf * 16 + j;
          pk[j] = (f16)((Ps[r * 64 + k] - mean) * inv * p.g[k] + p.b[k]);
        }
        *(v16h*)(Fout + (((kt * 4 + nt) * 32 + lr + 16 * hf) << 4)) = pk;
      }
  }
  __syncthreads();
}

__global__ __launch_bounds__(256) void score_kernel(
    const float* __restrict__ text_table, const float* __restrict__ image_table,
    const float* __restrict__ item_emb, ProjW tp, ProjW vp,
    const f16* __restrict__ querP, float* __restrict__ scores) {
  extern __shared__ char smem[];
  f16*   Xs = (f16*)smem;                          // 64KB
  f16*   Hs = (f16*)(smem + 64 * 1024);            // 16KB
  float* Ps = (float*)(smem + 80 * 1024);          // 16KB
  f16*   Fi = (f16*)(smem + 96 * 1024);            // 8KB (packed frags)
  f16*   Ft = (f16*)(smem + 104 * 1024);           // 8KB
  f16*   Fv = (f16*)(smem + 112 * 1024);           // 8KB

  int i0 = blockIdx.x * 64;
  int tid = threadIdx.x;
  int w = tid >> 5, lane = tid & 31, half = lane >> 4, nl = lane & 15;

  // warm L2 for the tiles we touch after the text projection
  for (int e = tid; e < 64 * 16; e += 256) {
    int row = e >> 4, c = e & 15;
    int item = min(i0 + row, NITEMS - 1);
    __builtin_prefetch(image_table + (size_t)item * 512 + c * 32, 0, 1);
  }
  if (tid < 128) {
    int row = tid >> 1, c = tid & 1;
    int item = min(i0 + row, NITEMS - 1);
    __builtin_prefetch(item_emb + (size_t)item * 64 + c * 32, 0, 1);
  }

  proj_tile(text_table, tp, i0, Xs, Hs, Ps, Ft);
  proj_tile(image_table, vp, i0, Xs, Hs, Ps, Fv);
  // item tile -> packed B fragments
  {
    int e = tid;  // 256 = 64 rows x {kt,hf}
    int r = e >> 2, kt = (e >> 1) & 1, hf = e & 1;
    int item = min(i0 + r, NITEMS - 1);
    const float* src = item_emb + (size_t)item * 64 + kt * 32 + hf * 16;
    v16h pk;
#pragma unroll
    for (int j = 0; j < 16; ++j) pk[j] = (f16)src[j];
    *(v16h*)(Fi + (((kt * 4 + (r >> 4)) * 32 + (r & 15) + 16 * hf) << 4)) = pk;
  }
  __syncthreads();

  // scores tile = Qi@Fi^T + Qt@Ft^T + Qv@Fv^T  (queries pre-scaled by mw/TEMP)
  const f16* Fm[3] = {Fi, Ft, Fv};
  for (int t = w; t < 16; t += 8) {
    int mt = t >> 2, nt = t & 3;
    v8f acc;
#pragma unroll
    for (int i = 0; i < 8; ++i) acc[i] = 0.0f;
#pragma unroll
    for (int kt = 0; kt < 2; ++kt) {
#pragma unroll
      for (int md = 0; md < 3; ++md) {
        v16h a = load_frag(querP, md * 8 + mt * 2 + kt, lane);
        v16h b = load_frag(Fm[md], kt * 4 + nt, lane);
        acc = wmma16(a, b, acc);
      }
    }
#pragma unroll
    for (int v = 0; v < 8; ++v) {
      int bidx = mt * 16 + v + 8 * half;
      int item = i0 + nt * 16 + nl;
      if (item < NITEMS) scores[(size_t)bidx * NITEMS + item] = acc[v];
    }
  }
}

// ------------------------------- top-k (20) --------------------------------

__global__ __launch_bounds__(256) void topk_kernel(const float* __restrict__ scores,
                                                   int* __restrict__ outidx) {
  int b = blockIdx.x, tid = threadIdx.x;
  float tv[20]; int ti[20];
#pragma unroll
  for (int j = 0; j < 20; ++j) { tv[j] = -1e30f; ti[j] = 0; }
  for (int i = tid; i < NITEMS; i += 256) {
    float v = scores[(size_t)b * NITEMS + i];
    if (v > tv[19]) {
      int j = 19;
      while (j > 0 && tv[j - 1] < v) { tv[j] = tv[j - 1]; ti[j] = ti[j - 1]; --j; }
      tv[j] = v; ti[j] = i;
    }
  }
  __shared__ float cv[256 * 20];
  __shared__ int   ci[256 * 20];
  __shared__ float rv[256];
  __shared__ int   ri[256];
#pragma unroll
  for (int j = 0; j < 20; ++j) { cv[tid * 20 + j] = tv[j]; ci[tid * 20 + j] = ti[j]; }
  __syncthreads();
  for (int sel = 0; sel < 20; ++sel) {
    float lm = -1e30f; int li = tid * 20;
    for (int j = 0; j < 20; ++j)
      if (cv[tid * 20 + j] > lm) { lm = cv[tid * 20 + j]; li = tid * 20 + j; }
    rv[tid] = lm; ri[tid] = li;
    __syncthreads();
    for (int st = 128; st > 0; st >>= 1) {
      if (tid < st && rv[tid + st] > rv[tid]) { rv[tid] = rv[tid + st]; ri[tid] = ri[tid + st]; }
      __syncthreads();
    }
    if (tid == 0) { outidx[b * 20 + sel] = ci[ri[0]]; cv[ri[0]] = -1e31f; }
    __syncthreads();
  }
}

// ------------------------------- host driver -------------------------------

extern "C" void kernel_launch(void* const* d_in, const int* in_sizes, int n_in,
                              void* d_out, int out_size, void* d_ws, size_t ws_size,
                              hipStream_t stream) {
  (void)in_sizes; (void)n_in; (void)out_size; (void)ws_size;
  auto F = [&](int i) { return (const float*)d_in[i]; };
  const int* hist = (const int*)d_in[0];
  const float* text_table = F(2);
  const float* image_table = F(3);
  // params flattened by JAX pytree order (dict keys sorted). Key indices:
  // 4-28 denoiser | 29 item_emb | 30-45 item_enc | 46-47 item_ln | 48 pos_emb
  // 49-52 router | 53-68 text_enc | 69-70 text_ln | 71-79 text_proj
  // 80-103 user_pref(item/text/visual) | 104-119 visual_enc | 120-121 visual_ln
  // 122-130 visual_proj

  // workspace layout (32B-aligned blocks)
  const size_t S = (size_t)BL * DD;
  float* w = (float*)d_ws;
  size_t off = 0;
  auto alloc  = [&](size_t n) { float* p = w + off; off += (n + 7) & ~(size_t)7; return p; };
  auto allocH = [&](size_t n) { f16* p = (f16*)(w + off); off += ((n + 15) & ~(size_t)15) / 2; return p; };
  int*   lens     = (int*)alloc(64);
  float* item_seq = alloc(S);
  float* text_seq = alloc(S);
  float* vis_seq  = alloc(S);
  float* t_de     = alloc(S);
  float* v_de     = alloc(S);
  float* iln      = alloc(S);
  float* tln      = alloc(S);
  float* vln      = alloc(S);
  float* item_en  = alloc(S);
  float* text_en  = alloc(S);
  float* vis_en   = alloc(S);
  float* t_de2    = alloc(S);
  float* v_de2    = alloc(S);
  float* tmpA = alloc(S); float* tmpB = alloc(S); float* tmpC = alloc(S);
  float* tmpD = alloc(S); float* tmpE = alloc(S); float* tmpF = alloc(S);
  float* tmpG = alloc(S);
  float* tmpH = alloc((size_t)BL * 256);
  float* gateH = alloc((size_t)BL * 32);
  float* gateB = alloc((size_t)BL * 2);
  float* prefP = alloc(3 * 64 * 64);
  float* poolP = alloc(3 * 64 * 64);
  float* mw    = alloc(64 * 3);
  f16*   querP = allocH(3 * 8 * 32 * 16);

  // packed weights
  auto pack = [&](const float* W, f16* Wp, int K, int N) {
    int groups = (K * N) >> 4;
    pack_b_kernel<<<(groups + 255) / 256, 256, 0, stream>>>(W, Wp, K, N);
  };
  f16* tw1p = allocH(512 * 128); f16* tw2p = allocH(128 * 64); f16* trsp = allocH(512 * 64);
  f16* vw1p = allocH(512 * 128); f16* vw2p = allocH(128 * 64); f16* vrsp = allocH(512 * 64);
  pack(F(77), tw1p, 512, 128); pack(F(79), tw2p, 128, 64); pack(F(75), trsp, 512, 64);
  pack(F(128), vw1p, 512, 128); pack(F(130), vw2p, 128, 64); pack(F(126), vrsp, 512, 64);
  // denoiser t2v/v2t q,k,v,o + gate.l1
  f16 *dQ[2], *dK[2], *dV[2], *dO[2];
  const int dq[2] = {18, 26}, dk[2] = {14, 22}, dv[2] = {20, 28}, dw[2] = {16, 24};
  for (int i = 0; i < 2; ++i) {
    dQ[i] = allocH(4096); dK[i] = allocH(4096); dV[i] = allocH(4096); dO[i] = allocH(4096);
    pack(F(dq[i]), dQ[i], 64, 64); pack(F(dk[i]), dK[i], 64, 64);
    pack(F(dv[i]), dV[i], 64, 64); pack(F(dw[i]), dO[i], 64, 64);
  }
  f16* gateP = allocH(64 * 32);
  pack(F(5), gateP, 64, 32);
  // encoders
  struct EncP { f16 *q, *k, *v, *o, *f1, *f2; } ep[3];
  const int eb[3] = {30, 53, 104};
  for (int i = 0; i < 3; ++i) {
    ep[i].q = allocH(4096); ep[i].k = allocH(4096); ep[i].v = allocH(4096);
    ep[i].o = allocH(4096); ep[i].f1 = allocH(64 * 256); ep[i].f2 = allocH(256 * 64);
    int e = eb[i];
    pack(F(e + 13), ep[i].q, 64, 64); pack(F(e + 9), ep[i].k, 64, 64);
    pack(F(e + 15), ep[i].v, 64, 64); pack(F(e + 11), ep[i].o, 64, 64);
    pack(F(e + 1), ep[i].f1, 64, 256); pack(F(e + 3), ep[i].f2, 256, 64);
  }
  // user_pref
  f16 *uQ[3], *uK[3], *uV[3], *uO[3];
  const int ub[3] = {80, 88, 96};
  for (int i = 0; i < 3; ++i) {
    uQ[i] = allocH(4096); uK[i] = allocH(4096); uV[i] = allocH(4096); uO[i] = allocH(4096);
    int u = ub[i];
    pack(F(u + 5), uQ[i], 64, 64); pack(F(u + 1), uK[i], 64, 64);
    pack(F(u + 7), uV[i], 64, 64); pack(F(u + 3), uO[i], 64, 64);
  }

  const int GB = BL / 16;  // 800 row-stripe blocks

  lens_kernel<<<1, 64, 0, stream>>>(hist, lens);
  gather_item_kernel<<<(BL * DD + 255) / 256, 256, 0, stream>>>(hist, F(29), item_seq);

  // ---- _proj on gathered sequence rows ----
  auto seq_proj = [&](const float* table, f16* w1p, f16* w2p, f16* rsp, int pb, float* out) {
    gemm_kernel<512, 128, 1><<<GB, 128, 0, stream>>>(table, hist, w1p, F(pb + 5), tmpH);
    gemm_kernel<128, 64, 0><<<GB, 128, 0, stream>>>(tmpH, nullptr, w2p, F(pb + 7), tmpA);
    gemm_kernel<512, 64, 0><<<GB, 128, 0, stream>>>(table, hist, rsp, F(pb + 3), tmpB);
    combine_ln_kernel<<<BL, 64, 0, stream>>>(out, tmpB, tmpA, F(pb + 0), nullptr, 0,
                                             nullptr, 0, F(pb + 2), F(pb + 1), 1e-5f);
  };
  seq_proj(text_table, tw1p, tw2p, trsp, 71, text_seq);
  seq_proj(image_table, vw1p, vw2p, vrsp, 122, vis_seq);

  // ---- denoiser ----
  auto denoiser = [&](const float* tin, const float* vin, const float* idin,
                      float* tout, float* vout) {
    gemm_kernel<64, 64, 0><<<GB, 128, 0, stream>>>(tin, nullptr, dQ[0], F(17), tmpA);
    gemm_kernel<64, 64, 0><<<GB, 128, 0, stream>>>(vin, nullptr, dK[0], F(13), tmpB);
    gemm_kernel<64, 64, 0><<<GB, 128, 0, stream>>>(vin, nullptr, dV[0], F(19), tmpC);
    attn_kernel<16, false><<<BB * 4, 256, 0, stream>>>(tmpA, tmpB, tmpC, lens, tmpD, 4);
    gemm_kernel<64, 64, 0><<<GB, 128, 0, stream>>>(tmpD, nullptr, dO[0], F(15), tmpF);
    gemm_kernel<64, 64, 0><<<GB, 128, 0, stream>>>(vin, nullptr, dQ[1], F(25), tmpA);
    gemm_kernel<64, 64, 0><<<GB, 128, 0, stream>>>(tin, nullptr, dK[1], F(21), tmpB);
    gemm_kernel<64, 64, 0><<<GB, 128, 0, stream>>>(tin, nullptr, dV[1], F(27), tmpC);
    attn_kernel<16, false><<<BB * 4, 256, 0, stream>>>(tmpA, tmpB, tmpC, lens, tmpD, 4);
    gemm_kernel<64, 64, 0><<<GB, 128, 0, stream>>>(tmpD, nullptr, dO[1], F(23), tmpG);
    gemm_kernel<64, 32, 1><<<GB, 128, 0, stream>>>(idin, nullptr, gateP, F(4), gateH);
    gate_kernel<<<(BL + 255) / 256, 256, 0, stream>>>(gateH, F(7), F(6), gateB);
    combine_ln_kernel<<<BL, 64, 0, stream>>>(tout, tin, tmpF, F(12), gateB, 0,
                                             nullptr, 0, F(9), F(8), 1e-5f);
    combine_ln_kernel<<<BL, 64, 0, stream>>>(vout, vin, tmpG, F(12), gateB, 1,
                                             nullptr, 0, F(11), F(10), 1e-5f);
  };
  denoiser(text_seq, vis_seq, item_seq, t_de, v_de);

  // ---- + pos, LN (eps 1e-12) ----
  combine_ln_kernel<<<BL, 64, 0, stream>>>(iln, item_seq, nullptr, nullptr, nullptr, 0,
                                           F(48), LL, F(47), F(46), 1e-12f);
  combine_ln_kernel<<<BL, 64, 0, stream>>>(tln, t_de, nullptr, nullptr, nullptr, 0,
                                           F(48), LL, F(70), F(69), 1e-12f);
  combine_ln_kernel<<<BL, 64, 0, stream>>>(vln, v_de, nullptr, nullptr, nullptr, 0,
                                           F(48), LL, F(121), F(120), 1e-12f);

  // ---- encoders ----
  auto encoder = [&](const float* xin, float* xout, int e, EncP& p) {
    gemm_kernel<64, 64, 0><<<GB, 128, 0, stream>>>(xin, nullptr, p.q, F(e + 12), tmpA);
    gemm_kernel<64, 64, 0><<<GB, 128, 0, stream>>>(xin, nullptr, p.k, F(e + 8), tmpB);
    gemm_kernel<64, 64, 0><<<GB, 128, 0, stream>>>(xin, nullptr, p.v, F(e + 14), tmpC);
    attn_kernel<8, true><<<BB * 8, 256, 0, stream>>>(tmpA, tmpB, tmpC, lens, tmpD, 8);
    gemm_kernel<64, 64, 0><<<GB, 128, 0, stream>>>(tmpD, nullptr, p.o, F(e + 10), tmpF);
    combine_ln_kernel<<<BL, 64, 0, stream>>>(tmpE, xin, tmpF, nullptr, nullptr, 0,
                                             nullptr, 0, F(e + 5), F(e + 4), 1e-5f);
    gemm_kernel<64, 256, 1><<<GB, 128, 0, stream>>>(tmpE, nullptr, p.f1, F(e + 0), tmpH);
    gemm_kernel<256, 64, 0><<<GB, 128, 0, stream>>>(tmpH, nullptr, p.f2, F(e + 2), tmpF);
    combine_ln_kernel<<<BL, 64, 0, stream>>>(xout, tmpE, tmpF, nullptr, nullptr, 0,
                                             nullptr, 0, F(e + 7), F(e + 6), 1e-5f);
  };
  encoder(iln, item_en, 30, ep[0]);
  encoder(tln, text_en, 53, ep[1]);
  encoder(vln, vis_en, 104, ep[2]);

  denoiser(text_en, vis_en, item_en, t_de2, v_de2);

  // ---- user-pref pooling ----
  auto user_pref = [&](const float* seq, int i, float* pref, float* pool) {
    int u = ub[i];
    gemm_kernel<64, 64, 0><<<GB, 128, 0, stream>>>(seq, nullptr, uQ[i], F(u + 4), tmpA);
    gemm_kernel<64, 64, 0><<<GB, 128, 0, stream>>>(seq, nullptr, uK[i], F(u + 0), tmpB);
    gemm_kernel<64, 64, 0><<<GB, 128, 0, stream>>>(seq, nullptr, uV[i], F(u + 6), tmpC);
    selfattend_kernel<<<BB, 256, 0, stream>>>(tmpA, tmpB, tmpC, lens, pref);
    gemm_kernel<64, 64, 0><<<4, 128, 0, stream>>>(pref, nullptr, uO[i], F(u + 2), pool);
  };
  user_pref(item_en, 0, prefP + 0 * 4096, poolP + 0 * 4096);
  user_pref(t_de2, 1, prefP + 1 * 4096, poolP + 1 * 4096);
  user_pref(v_de2, 2, prefP + 2 * 4096, poolP + 2 * 4096);

  router_kernel<<<1, 64, 0, stream>>>(poolP, poolP + 4096, poolP + 2 * 4096,
                                      F(50), F(49), F(52), F(51), mw);
  build_queries_kernel<<<3, 256, 0, stream>>>(mw, lens, item_en, t_de2, v_de2, querP);

  // ---- fused full-table projection + scoring ----
  ProjW tp = {tw1p, tw2p, trsp, F(76), F(78), F(74), F(71), F(73), F(72)};
  ProjW vp = {vw1p, vw2p, vrsp, F(127), F(129), F(125), F(122), F(124), F(123)};
  float* scores = (float*)d_out;
  score_kernel<<<(NITEMS + 63) / 64, 256, 120 * 1024, stream>>>(
      text_table, image_table, F(29), tp, vp, querP, scores);

  int* topk = (int*)(scores + (size_t)BB * NITEMS);
  topk_kernel<<<BB, 256, 0, stream>>>(scores, topk);
}
